// GCN_30253749633693
// MI455X (gfx1250) — matrix operand order repaired and checked by reading, not compiled, not run on hardware
//
#include <hip/hip_runtime.h>
#include <hip/hip_bf16.h>

typedef __attribute__((ext_vector_type(16))) _Float16 v16h;
typedef __attribute__((ext_vector_type(8)))  float    v8f;

// ---------------- utility kernels ----------------

__global__ void k_fill_f32(float* __restrict__ p, float v, int n) {
    int i = blockIdx.x * blockDim.x + threadIdx.x;
    if (i < n) p[i] = v;
}

__global__ void k_deg_count(const int* __restrict__ dst, float* __restrict__ deg, int E) {
    int e = blockIdx.x * blockDim.x + threadIdx.x;
    if (e < E) atomicAdd(&deg[dst[e]], 1.0f);
}

__global__ void k_rsqrt_inplace(float* __restrict__ p, int n) {
    int i = blockIdx.x * blockDim.x + threadIdx.x;
    if (i < n) p[i] = rsqrtf(p[i]);
}

__global__ void k_f32_to_f16(const float* __restrict__ in, _Float16* __restrict__ out, int n) {
    int i = blockIdx.x * blockDim.x + threadIdx.x;
    if (i < n) out[i] = (_Float16)in[i];
}

// ---------------- WMMA dense transform: hw[N,32] = h[N,fin] @ W16[fin,32] ----------------
// One wave32 computes a 16x32 output tile (two 16x16 WMMA N-tiles, shared A fragment).
// Requires row tile fully in range (N multiple of 16 handled; tail kernel covers remainder).

__global__ void k_gemm_wmma_f16(const float* __restrict__ h,
                                const _Float16* __restrict__ W16,
                                float* __restrict__ hw,
                                int ntiles, int fin) {
    int gtid = blockIdx.x * blockDim.x + threadIdx.x;
    int wave = gtid >> 5;
    if (wave >= ntiles) return;              // wave-uniform: EXEC stays all-ones inside
    int lane = threadIdx.x & 31;
    int m    = lane & 15;                    // row within tile (A/M) and N column (C/D)
    int hsel = lane >> 4;                    // half-wave select
    int row0 = wave << 4;

    const float* arow = h + (size_t)(row0 + m) * fin;
    v8f c0 = {};
    v8f c1 = {};

    for (int ks = 0; ks < fin; ks += 32) {
        // A fragment 16x32 f16: lane m holds K = ks + 8*hsel + {0..7} and ks + 16 + 8*hsel + {0..7}
        const float4* q0 = (const float4*)(arow + ks + 8 * hsel);
        const float4* q1 = (const float4*)(arow + ks + 16 + 8 * hsel);
        float4 u0 = q0[0], u1 = q0[1], u2 = q1[0], u3 = q1[1];
        v16h a;
        a[0]  = (_Float16)u0.x; a[1]  = (_Float16)u0.y; a[2]  = (_Float16)u0.z; a[3]  = (_Float16)u0.w;
        a[4]  = (_Float16)u1.x; a[5]  = (_Float16)u1.y; a[6]  = (_Float16)u1.z; a[7]  = (_Float16)u1.w;
        a[8]  = (_Float16)u2.x; a[9]  = (_Float16)u2.y; a[10] = (_Float16)u2.z; a[11] = (_Float16)u2.w;
        a[12] = (_Float16)u3.x; a[13] = (_Float16)u3.y; a[14] = (_Float16)u3.z; a[15] = (_Float16)u3.w;

        // B fragment 32x16 f16: lane holds column N = m, K in same order as A
        v16h b0, b1;
        for (int s = 0; s < 16; ++s) {
            int kk = ks + (s < 8 ? 8 * hsel + s : 16 + 8 * hsel + (s - 8));
            b0[s] = W16[(size_t)kk * 32 + m];
            b1[s] = W16[(size_t)kk * 32 + 16 + m];
        }

        c0 = __builtin_amdgcn_wmma_f32_16x16x32_f16(false, a, false, b0, (short)0, c0, false, false);
        c1 = __builtin_amdgcn_wmma_f32_16x16x32_f16(false, a, false, b1, (short)0, c1, false, false);
    }

    // C/D layout: VGPR r, lane L -> (M = r + 8*hsel, N = m)
    float* orow = hw + (size_t)row0 * 32;
#pragma unroll
    for (int r = 0; r < 8; ++r) {
        int mm = r + 8 * hsel;
        orow[(size_t)mm * 32 + m]      = c0[r];
        orow[(size_t)mm * 32 + 16 + m] = c1[r];
    }
}

// scalar tail for rows not covered by full 16-row WMMA tiles
__global__ void k_gemm_tail(const float* __restrict__ h, const _Float16* __restrict__ W16,
                            float* __restrict__ hw, int row_start, int nrows, int fin, int hid) {
    int t = blockIdx.x * blockDim.x + threadIdx.x;
    int nrem = (nrows - row_start) * hid;
    if (t >= nrem) return;
    int r = row_start + t / hid;
    int c = t % hid;
    float s = 0.0f;
    for (int k = 0; k < fin; ++k)
        s += h[(size_t)r * fin + k] * (float)W16[(size_t)k * hid + c];
    hw[(size_t)r * hid + c] = s;
}

// ---------------- edge propagation (HID == 32) ----------------

__global__ void k_selfloop_init(const float* __restrict__ hw, const float* __restrict__ dinv,
                                float* __restrict__ acc, int n) {
    int t = blockIdx.x * blockDim.x + threadIdx.x;
    if (t < n) {
        int i = t >> 5;
        float d = dinv[i];
        acc[t] = hw[t] * d * d;
    }
}

__global__ void k_edge_scatter(const float* __restrict__ hw, const float* __restrict__ dinv,
                               const int* __restrict__ src, const int* __restrict__ dst,
                               float* __restrict__ acc, int E) {
    int t = blockIdx.x * blockDim.x + threadIdx.x;
    int e = t >> 3;               // 8 lanes per edge, float4 each = 32 floats
    if (e >= E) return;
    int q = t & 7;
    int s = src[e];
    int d = dst[e];
    float w = dinv[s] * dinv[d];
    float4 v = *(const float4*)(hw + (size_t)s * 32 + q * 4);
    float* a = acc + (size_t)d * 32 + q * 4;
    atomicAdd(a + 0, v.x * w);
    atomicAdd(a + 1, v.y * w);
    atomicAdd(a + 2, v.z * w);
    atomicAdd(a + 3, v.w * w);
}

__global__ void k_bias_relu(const float* __restrict__ acc, const float* __restrict__ b,
                            float* __restrict__ hout, int n) {
    int t = blockIdx.x * blockDim.x + threadIdx.x;
    if (t < n) hout[t] = fmaxf(acc[t] + b[t & 31], 0.0f);
}

// ---------------- pooling + head ----------------

__global__ void k_pool_sum(const float* __restrict__ h, const int* __restrict__ batch,
                           float* __restrict__ gsum, int n) {
    int t = blockIdx.x * blockDim.x + threadIdx.x;
    if (t < n) {
        int i = t >> 5;
        atomicAdd(&gsum[(size_t)batch[i] * 32 + (t & 31)], h[t]);
    }
}

__global__ void k_pool_cnt(const int* __restrict__ batch, float* __restrict__ gcnt, int N) {
    int i = blockIdx.x * blockDim.x + threadIdx.x;
    if (i < N) atomicAdd(&gcnt[batch[i]], 1.0f);
}

__global__ void k_head(const float* __restrict__ gsum, const float* __restrict__ gcnt,
                       const float* __restrict__ Wl, const float* __restrict__ bl,
                       float* __restrict__ out, int G, int ncls) {
    int g = blockIdx.x * blockDim.x + threadIdx.x;
    if (g >= G) return;
    float pooled[32];
    float c = fmaxf(gcnt[g], 1.0f);
#pragma unroll
    for (int f = 0; f < 32; ++f) pooled[f] = gsum[(size_t)g * 32 + f] / c;
    float logit[16];
    float mx = -1e30f;
    for (int k = 0; k < ncls; ++k) {
        float s = bl[k];
        for (int f = 0; f < 32; ++f) s += pooled[f] * Wl[f * ncls + k];
        s = fmaxf(s, 0.0f);
        logit[k] = s;
        mx = fmaxf(mx, s);
    }
    float se = 0.0f;
    for (int k = 0; k < ncls; ++k) se += expf(logit[k] - mx);
    float lse = mx + logf(se);
    for (int k = 0; k < ncls; ++k) out[(size_t)g * ncls + k] = logit[k] - lse;
}

// ---------------- launcher ----------------

extern "C" void kernel_launch(void* const* d_in, const int* in_sizes, int n_in,
                              void* d_out, int out_size, void* d_ws, size_t ws_size,
                              hipStream_t stream) {
    const float* x    = (const float*)d_in[0];
    const int*   eidx = (const int*)d_in[1];
    const int*   bat  = (const int*)d_in[2];
    const float* W0   = (const float*)d_in[3];
    const float* b0   = (const float*)d_in[4];
    const float* W1   = (const float*)d_in[5];
    const float* b1   = (const float*)d_in[6];
    const float* W2   = (const float*)d_in[7];
    const float* b2   = (const float*)d_in[8];
    const float* Wl   = (const float*)d_in[9];
    const float* bl   = (const float*)d_in[10];

    const int Nn   = in_sizes[2];            // 100000 (batch has N elements)
    const int E    = in_sizes[1] / 2;        // 1600000
    const int fin0 = in_sizes[0] / Nn;       // 128
    const int HID  = in_sizes[4];            // 32
    const int ncls = in_sizes[10];           // 10
    const int G    = 128;

    const int* src  = eidx;
    const int* dstE = eidx + E;

    char* ws = (char*)d_ws;
    size_t off = 0;
    auto carve = [&](size_t bytes) -> char* {
        char* p = ws + off;
        off += (bytes + 255) & ~(size_t)255;
        return p;
    };
    float*    dinv  = (float*)carve((size_t)Nn * 4);
    float*    bufA  = (float*)carve((size_t)Nn * HID * 4);   // hw (GEMM out)
    float*    bufB  = (float*)carve((size_t)Nn * HID * 4);   // scatter accumulator
    float*    bufC  = (float*)carve((size_t)Nn * HID * 4);   // activation
    _Float16* W16_0 = (_Float16*)carve((size_t)fin0 * HID * 2);
    _Float16* W16_1 = (_Float16*)carve((size_t)HID * HID * 2);
    _Float16* W16_2 = (_Float16*)carve((size_t)HID * HID * 2);
    float*    gsum  = (float*)carve((size_t)G * HID * 4);
    float*    gcnt  = (float*)carve((size_t)G * 4);

    const int T = 256;
    auto nb = [](long long n, int t) { return (int)((n + t - 1) / t); };

    // degrees -> dinv (in place)
    k_fill_f32<<<nb(Nn, T), T, 0, stream>>>(dinv, 1.0f, Nn);           // self-loop
    k_deg_count<<<nb(E, T), T, 0, stream>>>(dstE, dinv, E);
    k_rsqrt_inplace<<<nb(Nn, T), T, 0, stream>>>(dinv, Nn);

    // weights -> f16 (row-major [K][32], matches WMMA B-fragment layout)
    k_f32_to_f16<<<nb((long long)fin0 * HID, T), T, 0, stream>>>(W0, W16_0, fin0 * HID);
    k_f32_to_f16<<<nb((long long)HID * HID, T), T, 0, stream>>>(W1, W16_1, HID * HID);
    k_f32_to_f16<<<nb((long long)HID * HID, T), T, 0, stream>>>(W2, W16_2, HID * HID);

    const float*    hin       = x;
    int             fin       = fin0;
    const float*    biases[3] = {b0, b1, b2};
    const _Float16* w16s[3]   = {W16_0, W16_1, W16_2};

    const int nNH   = Nn * HID;
    const int nfull = Nn & ~15;
    const int ntile = nfull >> 4;

    for (int layer = 0; layer < 3; ++layer) {
        k_gemm_wmma_f16<<<nb((long long)ntile * 32, T), T, 0, stream>>>(hin, w16s[layer], bufA, ntile, fin);
        if (Nn > nfull)
            k_gemm_tail<<<nb((long long)(Nn - nfull) * HID, T), T, 0, stream>>>(hin, w16s[layer], bufA,
                                                                                nfull, Nn, fin, HID);
        k_selfloop_init<<<nb(nNH, T), T, 0, stream>>>(bufA, dinv, bufB, nNH);
        k_edge_scatter<<<nb((long long)E * 8, T), T, 0, stream>>>(bufA, dinv, src, dstE, bufB, E);
        k_bias_relu<<<nb(nNH, T), T, 0, stream>>>(bufB, biases[layer], bufC, nNH);
        hin = bufC;
        fin = HID;
    }

    // global mean pool + head
    k_fill_f32<<<nb(G * HID, T), T, 0, stream>>>(gsum, 0.0f, G * HID);
    k_fill_f32<<<nb(G, T), T, 0, stream>>>(gcnt, 0.0f, G);
    k_pool_sum<<<nb(nNH, T), T, 0, stream>>>(bufC, bat, gsum, nNH);
    k_pool_cnt<<<nb(Nn, T), T, 0, stream>>>(bat, gcnt, Nn);
    k_head<<<1, G, 0, stream>>>(gsum, gcnt, Wl, bl, (float*)d_out, G, ncls);
}
